// Sender_18743237280009
// MI455X (gfx1250) — matrix operand order repaired and checked by reading, not compiled, not run on hardware
//
#include <hip/hip_runtime.h>

#define HC   128
#define EMB  64
#define NEGS 0.2f
#define CAP  8192

typedef __attribute__((ext_vector_type(16))) _Float16 v16h;
typedef __attribute__((ext_vector_type(8)))  _Float16 v8h;
typedef __attribute__((ext_vector_type(8)))  float    v8f;

__device__ __forceinline__ unsigned fkey(float f) {
    unsigned u = __float_as_uint(f);
    return (u & 0x80000000u) ? ~u : (u | 0x80000000u);
}
__device__ __forceinline__ float funkey(unsigned k) {
    unsigned u = (k & 0x80000000u) ? (k & 0x7fffffffu) : ~k;
    return __uint_as_float(u);
}
__device__ __forceinline__ float lrelu(float z) { return z > 0.f ? z : NEGS * z; }

// ---- sum(edge_attr) for the self-loop fill value (mean) ----
__global__ void k_reduce_sum(const float* __restrict__ x, float* __restrict__ out, int n) {
    float acc = 0.f;
    for (int i = blockIdx.x * blockDim.x + threadIdx.x; i < n; i += gridDim.x * blockDim.x)
        acc += x[i];
    for (int m = 16; m >= 1; m >>= 1) acc += __shfl_down(acc, m, 32);
    if ((threadIdx.x & 31) == 0) atomicAdd(out, acc);
}

// ---- conv1 pass 1: per-edge attention logits + running segment max ----
__global__ void k_edge_logits(const int* __restrict__ ei, const float* __restrict__ ea,
                              const float* __restrict__ x,
                              const float* __restrict__ Wl, const float* __restrict__ bl,
                              const float* __restrict__ Wr, const float* __restrict__ br,
                              const float* __restrict__ We, const float* __restrict__ att,
                              const float* __restrict__ meanSum,
                              float* __restrict__ exl, unsigned* __restrict__ mk,
                              int E, int Nn) {
    int e = blockIdx.x * blockDim.x + threadIdx.x;
    int total = E + Nn;
    if (e >= total) return;
    int src, dst; float a;
    if (e < E) { src = ei[e]; dst = ei[E + e]; a = ea[e]; }
    else       { src = e - E; dst = src;       a = meanSum[0] / (float)E; }
    float xs = x[src], xd = x[dst];
    float l0 = 0.f, l1 = 0.f;
#pragma unroll 8
    for (int c = 0; c < EMB; ++c) {
        float z = lrelu(xs * Wl[c] + bl[c] + xd * Wr[c] + br[c] + a * We[c]);
        l0 += att[c] * z;
    }
#pragma unroll 8
    for (int c = EMB; c < HC; ++c) {
        float z = lrelu(xs * Wl[c] + bl[c] + xd * Wr[c] + br[c] + a * We[c]);
        l1 += att[c] * z;
    }
    exl[e * 2 + 0] = l0;
    exl[e * 2 + 1] = l1;
    atomicMax(&mk[dst * 2 + 0], fkey(l0));
    atomicMax(&mk[dst * 2 + 1], fkey(l1));
}

// ---- conv1 pass 2: exp(logit - max), segment sums ----
__global__ void k_edge_norm(const int* __restrict__ ei, const unsigned* __restrict__ mk,
                            float* __restrict__ exl, float* __restrict__ s,
                            int E, int Nn) {
    int e = blockIdx.x * blockDim.x + threadIdx.x;
    int total = E + Nn;
    if (e >= total) return;
    int dst = (e < E) ? ei[E + e] : (e - E);
#pragma unroll
    for (int h = 0; h < 2; ++h) {
        float ex = __expf(exl[e * 2 + h] - funkey(mk[dst * 2 + h]));
        exl[e * 2 + h] = ex;
        atomicAdd(&s[dst * 2 + h], ex);
    }
}

// ---- conv1 pass 3: A[dst,h] += alpha * x[src]  (rank-1 message collapse) ----
__global__ void k_edge_acc(const int* __restrict__ ei, const float* __restrict__ x,
                           const float* __restrict__ exl, const float* __restrict__ s,
                           float* __restrict__ A, int E, int Nn) {
    int e = blockIdx.x * blockDim.x + threadIdx.x;
    int total = E + Nn;
    if (e >= total) return;
    int src, dst;
    if (e < E) { src = ei[e]; dst = ei[E + e]; }
    else       { src = e - E; dst = src; }
    float xs = x[src];
    float a0 = exl[e * 2 + 0] / s[dst * 2 + 0];
    float a1 = exl[e * 2 + 1] / s[dst * 2 + 1];
    atomicAdd(&A[dst * 2 + 0], a0 * xs);
    atomicAdd(&A[dst * 2 + 1], a1 * xs);
}

// ---- h1 = relu(Wl1[c]*A[i,head] + bl1[c] + b1[c]), stored f16 for the WMMA pipe ----
__global__ void k_h1(const float* __restrict__ A, const float* __restrict__ Wl,
                     const float* __restrict__ bl, const float* __restrict__ b1,
                     _Float16* __restrict__ h1h, int Nn) {
    int idx = blockIdx.x * blockDim.x + threadIdx.x;
    if (idx >= Nn * HC) return;
    int i = idx >> 7, c = idx & 127, h = c >> 6;
    float v = Wl[c] * A[i * 2 + h] + bl[c] + b1[c];
    h1h[idx] = (_Float16)(v > 0.f ? v : 0.f);
}

// ---- Wl2 [k][n] f32 -> col-major f16 Wt [n][k] so B fragments are contiguous ----
__global__ void k_prep_w(const float* __restrict__ W, _Float16* __restrict__ Wt) {
    int idx = blockIdx.x * blockDim.x + threadIdx.x;
    if (idx >= HC * HC) return;
    int k = idx >> 7, n = idx & 127;
    Wt[n * HC + k] = (_Float16)W[idx];
}

// ---- xl2 = h1 @ Wl2 + bl2.  Block = 8 waves = 128 rows; whole B staged in LDS (32KB);
//      each wave: 16x128 strip, A fragment reused across 8 WMMAs/k-step, 32 WMMAs total ----
__global__ void k_gemm_wmma(const _Float16* __restrict__ Ah, const _Float16* __restrict__ Wt,
                            const float* __restrict__ bias, float* __restrict__ Cm, int M) {
    __shared__ _Float16 Bs[HC * HC];
    {
        const v8h* gsrc = (const v8h*)Wt;
        v8h* ldst = (v8h*)Bs;
        for (int i = threadIdx.x; i < HC * HC / 8; i += blockDim.x) ldst[i] = gsrc[i];
    }
    __syncthreads();
    int wave = threadIdx.x >> 5;
    int lane = threadIdx.x & 31;
    int hl = lane >> 4, q = lane & 15;
    int rowBase = blockIdx.x * 128 + wave * 16;
    int row = rowBase + q; if (row >= M) row = M - 1;
    const _Float16* arow = Ah + (size_t)row * HC;
    v8f acc[8];
#pragma unroll
    for (int t = 0; t < 8; ++t) acc[t] = (v8f){};
#pragma unroll
    for (int k0 = 0; k0 < HC; k0 += 32) {
        // A 16x32 f16 fragment: lane = row, K = k0 + 16*half + j, contiguous 16xf16
        v16h a = *(const v16h*)(arow + k0 + 16 * hl);
#pragma unroll
        for (int tn = 0; tn < 8; ++tn) {
            // B 32x16 fragment: lane = col, K = k0 + 16*half + j, contiguous in Bs
            v16h b = *(const v16h*)(Bs + (tn * 16 + q) * HC + k0 + 16 * hl);
            acc[tn] = __builtin_amdgcn_wmma_f32_16x16x32_f16(false, a, false, b,
                                                             (short)0, acc[tn], false, false);
        }
    }
#pragma unroll
    for (int tn = 0; tn < 8; ++tn) {
        float bv = bias[tn * 16 + q];
#pragma unroll
        for (int r = 0; r < 8; ++r) {    // D: VGPR r -> M = r + 8*half, N = lane%16
            int grow = rowBase + r + (hl << 3);
            if (grow < M) Cm[(size_t)grow * HC + tn * 16 + q] = acc[tn][r] + bv;
        }
    }
}

// ---- xr2 needed only at target row: 128-thread GEMV ----
__global__ void k_row_gemv(const _Float16* __restrict__ h1h, const float* __restrict__ W,
                           const float* __restrict__ bias, const int* __restrict__ tgt,
                           float* __restrict__ out128) {
    int c = threadIdx.x;
    const _Float16* hr = h1h + (size_t)tgt[0] * HC;
    float acc = bias[c];
    for (int k = 0; k < HC; ++k) acc += (float)hr[k] * W[k * HC + c];
    out128[c] = acc;
}

// ---- conv2 edge phase restricted to dst == target: compact matches + logits ----
__global__ void k_scan_target(const int* __restrict__ ei, const float* __restrict__ ea,
                              const float* __restrict__ meanSum,
                              const float* __restrict__ xl2, const float* __restrict__ xr2t,
                              const float* __restrict__ We, const float* __restrict__ att,
                              const int* __restrict__ tgt, int* __restrict__ cnt,
                              int* __restrict__ csrc, float* __restrict__ clog,
                              int E, int Nn) {
    int e = blockIdx.x * blockDim.x + threadIdx.x;
    int total = E + Nn;
    if (e >= total) return;
    int target = tgt[0];
    int src, dst; float a;
    if (e < E) { src = ei[e]; dst = ei[E + e]; a = ea[e]; }
    else       { src = e - E; dst = src;       a = meanSum[0] / (float)E; }
    if (dst != target) return;
    const float* xs = xl2 + (size_t)src * HC;
    float l0 = 0.f, l1 = 0.f;
    for (int c = 0; c < EMB; ++c) {
        float z = lrelu(xs[c] + xr2t[c] + a * We[c]);
        l0 += att[c] * z;
    }
    for (int c = EMB; c < HC; ++c) {
        float z = lrelu(xs[c] + xr2t[c] + a * We[c]);
        l1 += att[c] * z;
    }
    int pos = atomicAdd(cnt, 1);
    if (pos < CAP) { csrc[pos] = src; clog[pos * 2] = l0; clog[pos * 2 + 1] = l1; }
}

// ---- softmax over target edges, aggregate, add b2, then fc(128x128) ----
__global__ void k_finalize(const int* __restrict__ cnt, const int* __restrict__ csrc,
                           const float* __restrict__ clog, const float* __restrict__ xl2,
                           const float* __restrict__ b2, const float* __restrict__ Wfc,
                           const float* __restrict__ bfc, float* __restrict__ out) {
    __shared__ float h2[HC];
    int c = threadIdx.x;            // 0..127
    int n = cnt[0]; if (n > CAP) n = CAP;
    int h = c >> 6;
    float m = -3.4e38f;
    for (int i = 0; i < n; ++i) m = fmaxf(m, clog[i * 2 + h]);
    float s = 0.f;
    for (int i = 0; i < n; ++i) s += __expf(clog[i * 2 + h] - m);
    float acc = 0.f;
    for (int i = 0; i < n; ++i) {
        float alpha = __expf(clog[i * 2 + h] - m) / s;
        acc += alpha * xl2[(size_t)csrc[i] * HC + c];
    }
    h2[c] = acc + b2[c];
    __syncthreads();
    float o = bfc[c];
    for (int k = 0; k < HC; ++k) o += h2[k] * Wfc[k * HC + c];
    out[c] = o;
}

extern "C" void kernel_launch(void* const* d_in, const int* in_sizes, int n_in,
                              void* d_out, int out_size, void* d_ws, size_t ws_size,
                              hipStream_t stream) {
    const float* x    = (const float*)d_in[0];
    const int*   ei   = (const int*)d_in[1];
    const float* ea   = (const float*)d_in[2];
    const int*   tgt  = (const int*)d_in[3];
    const float* Wl1  = (const float*)d_in[4];
    const float* bl1  = (const float*)d_in[5];
    const float* Wr1  = (const float*)d_in[6];
    const float* br1  = (const float*)d_in[7];
    const float* We1  = (const float*)d_in[8];
    const float* att1 = (const float*)d_in[9];
    const float* b1   = (const float*)d_in[10];
    const float* Wl2  = (const float*)d_in[11];
    const float* bl2  = (const float*)d_in[12];
    const float* Wr2  = (const float*)d_in[13];
    const float* br2  = (const float*)d_in[14];
    const float* We2  = (const float*)d_in[15];
    const float* att2 = (const float*)d_in[16];
    const float* b2   = (const float*)d_in[17];
    const float* Wfc  = (const float*)d_in[18];
    const float* bfc  = (const float*)d_in[19];

    int Nn = in_sizes[0];
    int E  = in_sizes[1] / 2;
    int total = E + Nn;
    (void)n_in; (void)out_size; (void)ws_size;

    char* w = (char*)d_ws;
    size_t off = 0;
    auto carve = [&](size_t bytes) -> char* {
        char* p = w + off;
        off = (off + bytes + 255) & ~(size_t)255;
        return p;
    };
    _Float16* h1h = (_Float16*)carve((size_t)Nn * HC * 2);
    _Float16* Wt  = (_Float16*)carve((size_t)HC * HC * 2);
    float*    xl2 = (float*)carve((size_t)Nn * HC * 4);
    float*    exl = (float*)carve((size_t)total * 2 * 4);
    size_t zbytes = (size_t)Nn * 2 * 4 * 3 + 8;   // s, A, mkeys, meanSum, cnt
    char* zp = carve(zbytes);
    float*    s_      = (float*)zp;
    float*    A_      = s_ + 2 * (size_t)Nn;
    unsigned* mk      = (unsigned*)(A_ + 2 * (size_t)Nn);
    float*    meanSum = (float*)(mk + 2 * (size_t)Nn);
    int*      cnt     = (int*)(meanSum + 1);
    float* xr2t = (float*)carve(HC * 4);
    int*   csrc = (int*)carve(CAP * 4);
    float* clog = (float*)carve((size_t)CAP * 2 * 4);

    hipMemsetAsync(zp, 0, zbytes, stream);

    const int TB = 256;
    int gEdges = (total + TB - 1) / TB;

    k_reduce_sum<<<256, TB, 0, stream>>>(ea, meanSum, E);
    k_prep_w<<<(HC * HC + TB - 1) / TB, TB, 0, stream>>>(Wl2, Wt);
    k_edge_logits<<<gEdges, TB, 0, stream>>>(ei, ea, x, Wl1, bl1, Wr1, br1, We1, att1,
                                             meanSum, exl, mk, E, Nn);
    k_edge_norm<<<gEdges, TB, 0, stream>>>(ei, mk, exl, s_, E, Nn);
    k_edge_acc<<<gEdges, TB, 0, stream>>>(ei, x, exl, s_, A_, E, Nn);

    int gH = (Nn * HC + TB - 1) / TB;
    k_h1<<<gH, TB, 0, stream>>>(A_, Wl1, bl1, b1, h1h, Nn);

    int gG = (Nn + 127) / 128;
    k_gemm_wmma<<<gG, TB, 0, stream>>>(h1h, Wt, bl2, xl2, Nn);
    k_row_gemv<<<1, HC, 0, stream>>>(h1h, Wr2, br2, tgt, xr2t);

    k_scan_target<<<gEdges, TB, 0, stream>>>(ei, ea, meanSum, xl2, xr2t, We2, att2,
                                             tgt, cnt, csrc, clog, E, Nn);
    k_finalize<<<1, HC, 0, stream>>>(cnt, csrc, clog, xl2, b2, Wfc, bfc, (float*)d_out);
}
